// DiscretePredictor_84430467104804
// MI455X (gfx1250) — compile-verified, matrix operand order checked by reading, not run on hardware
//
#include <hip/hip_runtime.h>

typedef __bf16 bf16;
typedef __attribute__((ext_vector_type(16))) __bf16 v16bf;
typedef __attribute__((ext_vector_type(8)))  __bf16 v8bf;
typedef __attribute__((ext_vector_type(8)))  float  v8f;

#define ROWS  8192      // BATCH * NUM_OBJ
#define DIM   128
#define HID   256
#define NPAIR 262144.0  // BATCH * 32 * 32

__device__ __forceinline__ float leakyf(float x) { return x >= 0.f ? x : 0.01f * x; }

// ---------------------------------------------------------------------------
// WMMA helpers: bf16 16x16x32, f32 accumulate. NT fragment loader: for both
// A (MxK row-major) and B (NxK row-major, i.e. W for x@W^T), lane L holds
// row r = L%16, and K elements {k+8*half..k+8*half+7, k+16+8*half..+7}.
// ---------------------------------------------------------------------------
__device__ __forceinline__ v16bf load_frag(const bf16* __restrict__ base, int ld,
                                           int row, int kbase, int half) {
  const bf16* p = base + (size_t)row * ld + kbase + half * 8;
  v8bf a = *(const v8bf*)p;
  v8bf b = *(const v8bf*)(p + 16);
  v16bf f;
#pragma unroll
  for (int i = 0; i < 8; ++i) { f[i] = a[i]; f[i + 8] = b[i]; }
  return f;
}

__device__ __forceinline__ v8f wmma_acc(v16bf a, v16bf b, v8f c) {
  return __builtin_amdgcn_wmma_f32_16x16x32_bf16(false, a, false, b, (short)0, c,
                                                 false, false);
}

// ---------------------------------------------------------------------------
// C[MxN] = A[MxK] * B[NxK]^T + bias_scale*bias[n], with A,B as bf16 hi/lo
// pairs (3-WMMA error-compensated bf16 -> ~f32 accuracy).
// Wave tile 16x32, workgroup (8 waves) tile 64x64. grid=(M/64, N/64).
// ---------------------------------------------------------------------------
__global__ __launch_bounds__(256) void gemm_nt_bf16x3(
    const bf16* __restrict__ Ahi, const bf16* __restrict__ Alo, int lda,
    const bf16* __restrict__ Bhi, const bf16* __restrict__ Blo, int ldb,
    float* __restrict__ C, int ldc,
    const float* __restrict__ bias, float bias_scale, int K) {
  int lane = threadIdx.x & 31;
  int wave = threadIdx.x >> 5;
  int m0 = blockIdx.x * 64 + (wave & 3) * 16;
  int n0 = blockIdx.y * 64 + (wave >> 2) * 32;
  int r = lane & 15;
  int half = lane >> 4;

  v8f c0 = {}; v8f c1 = {};
  for (int k = 0; k < K; k += 32) {
    v16bf ah = load_frag(Ahi, lda, m0 + r, k, half);
    v16bf al = load_frag(Alo, lda, m0 + r, k, half);
    v16bf bh0 = load_frag(Bhi, ldb, n0 + r, k, half);
    v16bf bl0 = load_frag(Blo, ldb, n0 + r, k, half);
    v16bf bh1 = load_frag(Bhi, ldb, n0 + 16 + r, k, half);
    v16bf bl1 = load_frag(Blo, ldb, n0 + 16 + r, k, half);
    c0 = wmma_acc(ah, bh0, c0);
    c0 = wmma_acc(ah, bl0, c0);
    c0 = wmma_acc(al, bh0, c0);
    c1 = wmma_acc(ah, bh1, c1);
    c1 = wmma_acc(ah, bl1, c1);
    c1 = wmma_acc(al, bh1, c1);
  }
  // C/D layout: vgpr v, lane -> C[m0 + v + 8*half][n0 + lane%16]
  float bv0 = bias ? bias_scale * bias[n0 + r] : 0.f;
  float bv1 = bias ? bias_scale * bias[n0 + 16 + r] : 0.f;
#pragma unroll
  for (int v = 0; v < 8; ++v) {
    int m = m0 + v + half * 8;
    C[(size_t)m * ldc + n0 + r]      = c0[v] + bv0;
    C[(size_t)m * ldc + n0 + 16 + r] = c1[v] + bv1;
  }
}

// f32 -> (bf16 hi, bf16 lo) split
__global__ void cvt_hilo(const float* __restrict__ x, bf16* __restrict__ hi,
                         bf16* __restrict__ lo, int n) {
  int i = blockIdx.x * 256 + threadIdx.x;
  if (i < n) {
    float v = x[i];
    bf16 h = (bf16)v;
    hi[i] = h;
    lo[i] = (bf16)(v - (float)h);
  }
}

// Per batch b: row/col degrees of E_b (32x32) and EQ[b,i,c] = sum_j e_ij Q[b,j,c]
__global__ __launch_bounds__(256) void batch_prep(
    const float* __restrict__ edges, const float* __restrict__ Q,
    float* __restrict__ rowdeg, float* __restrict__ coldeg, float* __restrict__ EQ) {
  __shared__ float se[1024];
  __shared__ float sq[32 * 256];
  int b = blockIdx.x, t = threadIdx.x;
  for (int i = t; i < 1024; i += 256) se[i] = edges[b * 1024 + i];
  for (int i = t; i < 32 * 256; i += 256) sq[i] = Q[(size_t)b * 32 * 256 + i];
  __syncthreads();
  if (t < 32) {
    float s = 0.f;
    for (int j = 0; j < 32; ++j) s += se[t * 32 + j];
    rowdeg[b * 32 + t] = s;
  } else if (t < 64) {
    int j = t - 32;
    float s = 0.f;
    for (int i = 0; i < 32; ++i) s += se[i * 32 + j];
    coldeg[b * 32 + j] = s;
  }
  int c = t;
  for (int i = 0; i < 32; ++i) {
    float acc = 0.f;
    for (int j = 0; j < 32; ++j) acc += se[i * 32 + j] * sq[j * 256 + c];
    EQ[(size_t)(b * 32 + i) * 256 + c] = acc;
  }
}

// Analytic BN stats for the (never-materialized) 262144x256 hidden matrix.
// stage1: grid 64, 128 rows each, thread = channel. Double accumulators.
__global__ __launch_bounds__(256) void msg_stats1(
    const float* __restrict__ P, const float* __restrict__ Q,
    const float* __restrict__ EQ, const float* __restrict__ rowdeg,
    const float* __restrict__ coldeg, double* __restrict__ partial) {
  int w = blockIdx.x, c = threadIdx.x;
  double s1 = 0, s2 = 0, s3 = 0, s4 = 0, s5 = 0;
  for (int r = w * 128; r < w * 128 + 128; ++r) {
    double rd = rowdeg[r], cd = coldeg[r];
    double p = P[(size_t)r * 256 + c], q = Q[(size_t)r * 256 + c];
    double eq = EQ[(size_t)r * 256 + c];
    s1 += rd * p;  s3 += rd * p * p;
    s2 += cd * q;  s4 += cd * q * q;
    s5 += p * eq;
  }
  partial[(size_t)(w * 5 + 0) * 256 + c] = s1;
  partial[(size_t)(w * 5 + 1) * 256 + c] = s2;
  partial[(size_t)(w * 5 + 2) * 256 + c] = s3;
  partial[(size_t)(w * 5 + 3) * 256 + c] = s4;
  partial[(size_t)(w * 5 + 4) * 256 + c] = s5;
}

// stage2: combine -> per-channel affine coeffs. h~ = leaky(a*(P+Q) + cB) for e=1,
// and leaky(cB) for e=0. coef = [a | cB | leaky(cB)]
__global__ void msg_stats2(const double* __restrict__ partial,
                           const float* __restrict__ b1,
                           const float* __restrict__ gamma,
                           const float* __restrict__ beta,
                           float* __restrict__ coef) {
  int c = threadIdx.x;
  double s1 = 0, s2 = 0, s3 = 0, s4 = 0, s5 = 0;
  for (int w = 0; w < 64; ++w) {
    s1 += partial[(size_t)(w * 5 + 0) * 256 + c];
    s2 += partial[(size_t)(w * 5 + 1) * 256 + c];
    s3 += partial[(size_t)(w * 5 + 2) * 256 + c];
    s4 += partial[(size_t)(w * 5 + 3) * 256 + c];
    s5 += partial[(size_t)(w * 5 + 4) * 256 + c];
  }
  double bb = (double)b1[c];
  double Sh = s1 + s2 + NPAIR * bb;
  double mean = Sh / NPAIR;
  double Sh2 = s3 + s4 + 2.0 * s5 + 2.0 * bb * (s1 + s2) + NPAIR * bb * bb;
  double var = Sh2 / NPAIR - mean * mean;
  float inv = rsqrtf((float)var + 1e-5f);
  float a = gamma[c] * inv;
  float cB = a * (float)bb + beta[c] - a * (float)mean;
  coef[c] = a;
  coef[256 + c] = cB;
  coef[512 + c] = leakyf(cB);
}

// Per-pair leaky-affine + sum over j, output S = sum_j h~ as bf16 hi/lo.
// One workgroup per batch; e is uniform across the wave -> scalar branch.
__global__ __launch_bounds__(256) void msg_pass(
    const float* __restrict__ P, const float* __restrict__ Q,
    const float* __restrict__ edges, const float* __restrict__ rowdeg,
    const float* __restrict__ coef, bf16* __restrict__ Shi, bf16* __restrict__ Slo) {
  __shared__ float aq[32 * 256];
  __shared__ float se[1024];
  int b = blockIdx.x, t = threadIdx.x;
  float a = coef[t], cB = coef[256 + t], h0 = coef[512 + t];
  for (int i = t; i < 1024; i += 256) se[i] = edges[b * 1024 + i];
  for (int idx = t; idx < 32 * 256; idx += 256)  // c == t for stride-256 loop
    aq[idx] = a * Q[(size_t)b * 8192 + idx];
  __syncthreads();
  int c = t;
  for (int i = 0; i < 32; ++i) {
    int r = b * 32 + i;
    float api = a * P[(size_t)r * 256 + c] + cB;
    float acc = (32.f - rowdeg[r]) * h0;
    for (int j = 0; j < 32; ++j) {
      if (se[i * 32 + j] != 0.f) {
        float x = api + aq[j * 256 + c];
        acc += leakyf(x);
      }
    }
    bf16 h = (bf16)acc;
    size_t off = (size_t)r * 256 + c;
    Shi[off] = h;
    Slo[off] = (bf16)(acc - (float)h);
  }
}

// sm = concat[state, agg] -> bf16 hi/lo (8192 x 256)
__global__ void build_sm(const float* __restrict__ state, const float* __restrict__ agg,
                         bf16* __restrict__ hi, bf16* __restrict__ lo) {
  int idx = blockIdx.x * 256 + threadIdx.x;
  int r = idx >> 8, c = idx & 255;
  float v = (c < 128) ? state[(size_t)r * 128 + c] : agg[(size_t)r * 128 + (c - 128)];
  bf16 h = (bf16)v;
  hi[idx] = h;
  lo[idx] = (bf16)(v - (float)h);
}

// Standard 2-moment BN stats over H (8192 x 256), two stages, double acc.
__global__ __launch_bounds__(256) void fin_stats1(const float* __restrict__ X,
                                                  double* __restrict__ partial) {
  int w = blockIdx.x, c = threadIdx.x;
  double s = 0, s2 = 0;
  for (int r = w * 128; r < w * 128 + 128; ++r) {
    double x = X[(size_t)r * 256 + c];
    s += x; s2 += x * x;
  }
  partial[(size_t)(w * 2 + 0) * 256 + c] = s;
  partial[(size_t)(w * 2 + 1) * 256 + c] = s2;
}

__global__ void fin_stats2(const double* __restrict__ partial,
                           const float* __restrict__ gamma,
                           const float* __restrict__ beta, float* __restrict__ coef) {
  int c = threadIdx.x;
  double s = 0, s2 = 0;
  for (int w = 0; w < 64; ++w) {
    s += partial[(size_t)(w * 2 + 0) * 256 + c];
    s2 += partial[(size_t)(w * 2 + 1) * 256 + c];
  }
  double N = 8192.0;
  double mean = s / N;
  double var = s2 / N - mean * mean;
  float inv = rsqrtf((float)var + 1e-5f);
  float a = gamma[c] * inv;
  coef[c] = a;
  coef[256 + c] = beta[c] - a * (float)mean;
}

// H~ = leaky(a*H + d) -> bf16 hi/lo
__global__ void act_hilo(const float* __restrict__ H, const float* __restrict__ coef,
                         bf16* __restrict__ hi, bf16* __restrict__ lo) {
  int idx = blockIdx.x * 256 + threadIdx.x;
  int c = idx & 255;
  float x = coef[c] * H[idx] + coef[256 + c];
  x = leakyf(x);
  bf16 h = (bf16)x;
  hi[idx] = h;
  lo[idx] = (bf16)(x - (float)h);
}

extern "C" void kernel_launch(void* const* d_in, const int* in_sizes, int n_in,
                              void* d_out, int out_size, void* d_ws, size_t ws_size,
                              hipStream_t stream) {
  (void)in_sizes; (void)n_in; (void)out_size; (void)ws_size;
  const float* state     = (const float*)d_in[0];
  const float* edges     = (const float*)d_in[1];
  const float* msg_w1    = (const float*)d_in[2];
  const float* msg_b1    = (const float*)d_in[3];
  const float* msg_gamma = (const float*)d_in[4];
  const float* msg_beta  = (const float*)d_in[5];
  const float* msg_w2    = (const float*)d_in[6];
  const float* msg_b2    = (const float*)d_in[7];
  const float* fin_w1    = (const float*)d_in[8];
  const float* fin_b1    = (const float*)d_in[9];
  const float* fin_gamma = (const float*)d_in[10];
  const float* fin_beta  = (const float*)d_in[11];
  const float* fin_w2    = (const float*)d_in[12];
  const float* fin_b2    = (const float*)d_in[13];

  char* ws = (char*)d_ws;
  size_t o = 0;
  auto alloc = [&](size_t sz) -> char* {
    size_t r = o;
    o = (o + sz + 255) & ~(size_t)255;
    return ws + r;
  };
  bf16* state_hi = (bf16*)alloc((size_t)ROWS * DIM * 2);
  bf16* state_lo = (bf16*)alloc((size_t)ROWS * DIM * 2);
  bf16* w1_hi = (bf16*)alloc(256 * 256 * 2);
  bf16* w1_lo = (bf16*)alloc(256 * 256 * 2);
  bf16* w2_hi = (bf16*)alloc(128 * 256 * 2);
  bf16* w2_lo = (bf16*)alloc(128 * 256 * 2);
  bf16* f1_hi = (bf16*)alloc(256 * 256 * 2);
  bf16* f1_lo = (bf16*)alloc(256 * 256 * 2);
  bf16* f2_hi = (bf16*)alloc(128 * 256 * 2);
  bf16* f2_lo = (bf16*)alloc(128 * 256 * 2);
  float* P  = (float*)alloc((size_t)ROWS * HID * 4);
  float* Q  = (float*)alloc((size_t)ROWS * HID * 4);
  float* EQ = (float*)alloc((size_t)ROWS * HID * 4);
  float* rowdeg = (float*)alloc(ROWS * 4);
  float* coldeg = (float*)alloc(ROWS * 4);
  double* partial = (double*)alloc((size_t)64 * 5 * 256 * 8);
  float* msgcoef = (float*)alloc(3 * 256 * 4);
  float* fincoef = (float*)alloc(2 * 256 * 4);
  bf16* Shi = (bf16*)alloc((size_t)ROWS * HID * 2);
  bf16* Slo = (bf16*)alloc((size_t)ROWS * HID * 2);
  float* agg = (float*)alloc((size_t)ROWS * DIM * 4);
  // lifetime-safe buffer reuse:
  float* H = P;                                              // P dead after msg_pass
  bf16* sm_hi = (bf16*)Q;                                    // Q dead after msg_pass
  bf16* sm_lo = (bf16*)((char*)Q + (size_t)ROWS * HID * 2);
  bf16* ht_hi = (bf16*)EQ;                                   // EQ dead after msg_stats1
  bf16* ht_lo = (bf16*)((char*)EQ + (size_t)ROWS * HID * 2);

  // precision split of inputs/weights
  cvt_hilo<<<(ROWS * DIM + 255) / 256, 256, 0, stream>>>(state, state_hi, state_lo, ROWS * DIM);
  cvt_hilo<<<(65536 + 255) / 256, 256, 0, stream>>>(msg_w1, w1_hi, w1_lo, 65536);
  cvt_hilo<<<(32768 + 255) / 256, 256, 0, stream>>>(msg_w2, w2_hi, w2_lo, 32768);
  cvt_hilo<<<(65536 + 255) / 256, 256, 0, stream>>>(fin_w1, f1_hi, f1_lo, 65536);
  cvt_hilo<<<(32768 + 255) / 256, 256, 0, stream>>>(fin_w2, f2_hi, f2_lo, 32768);

  // P = state @ W1a^T, Q = state @ W1b^T   (8192x256x128 each)
  dim3 gPQ(ROWS / 64, HID / 64);
  gemm_nt_bf16x3<<<gPQ, 256, 0, stream>>>(state_hi, state_lo, DIM, w1_hi, w1_lo, 256,
                                          P, HID, nullptr, 0.f, DIM);
  gemm_nt_bf16x3<<<gPQ, 256, 0, stream>>>(state_hi, state_lo, DIM, w1_hi + DIM,
                                          w1_lo + DIM, 256, Q, HID, nullptr, 0.f, DIM);

  // analytic BN stats for message layer
  batch_prep<<<256, 256, 0, stream>>>(edges, Q, rowdeg, coldeg, EQ);
  msg_stats1<<<64, 256, 0, stream>>>(P, Q, EQ, rowdeg, coldeg, partial);
  msg_stats2<<<1, 256, 0, stream>>>(partial, msg_b1, msg_gamma, msg_beta, msgcoef);

  // S[b,i] = sum_j leaky(BN(h))  (the only per-pair work)
  msg_pass<<<256, 256, 0, stream>>>(P, Q, edges, rowdeg, msgcoef, Shi, Slo);

  // agg = S @ W2^T + 32*b2   (8192x128x256)
  dim3 gA(ROWS / 64, DIM / 64);
  gemm_nt_bf16x3<<<gA, 256, 0, stream>>>(Shi, Slo, HID, w2_hi, w2_lo, 256,
                                         agg, DIM, msg_b2, 32.f, HID);

  // final MLP
  build_sm<<<(ROWS * HID + 255) / 256, 256, 0, stream>>>(state, agg, sm_hi, sm_lo);
  dim3 gH(ROWS / 64, HID / 64);
  gemm_nt_bf16x3<<<gH, 256, 0, stream>>>(sm_hi, sm_lo, HID, f1_hi, f1_lo, 256,
                                         H, HID, fin_b1, 1.f, HID);
  fin_stats1<<<64, 256, 0, stream>>>(H, partial);
  fin_stats2<<<1, 256, 0, stream>>>(partial, fin_gamma, fin_beta, fincoef);
  act_hilo<<<(ROWS * HID + 255) / 256, 256, 0, stream>>>(H, fincoef, ht_hi, ht_lo);

  dim3 gO(ROWS / 64, DIM / 64);
  gemm_nt_bf16x3<<<gO, 256, 0, stream>>>(ht_hi, ht_lo, HID, f2_hi, f2_lo, 256,
                                         (float*)d_out, DIM, fin_b2, 1.f, HID);
}
